// SelfAttention_80032420593800
// MI455X (gfx1250) — matrix operand order repaired and checked by reading, not compiled, not run on hardware
//
#include <hip/hip_runtime.h>
#include <hip/hip_bf16.h>
#include <math.h>

typedef __attribute__((ext_vector_type(16))) _Float16 v16h;
typedef __attribute__((ext_vector_type(8)))  _Float16 v8h;
typedef __attribute__((ext_vector_type(8)))  float    v8f;

#define LSEQ   2048
#define DHID   2048
#define NBATCH 2
#define NHEAD  16
#define DHEAD  128
#define MTOT   (NBATCH * LSEQ)   // 4096 rows in all projection GEMMs

// ---------------------------------------------------------------------------
// WMMA helpers (layouts per CDNA5 ISA 7.12.2, wave32)
// ---------------------------------------------------------------------------
__device__ __forceinline__ v8f wmma_f16(v16h a, v16h b, v8f c) {
  return __builtin_amdgcn_wmma_f32_16x16x32_f16(
      /*neg_a=*/false, a, /*neg_b=*/false, b,
      /*c_mod=*/(short)0, c, /*reuse_a=*/false, /*reuse_b=*/false);
}

// A fragment: 16x32 (MxK) f16, row-major source with leading dim ld.
// lane holds row m0+(lane&15); 16 contiguous K values [h*16, h*16+16).
__device__ __forceinline__ v16h load_A(const _Float16* __restrict__ p, int ld,
                                       int m0, int k0, int lane) {
  const int m = m0 + (lane & 15);
  const int h = lane >> 4;
  const _Float16* q = p + (size_t)m * ld + k0 + h * 16;
  v8h lo = *(const v8h*)(q);
  v8h hi = *(const v8h*)(q + 8);
  return __builtin_shufflevector(lo, hi, 0,1,2,3,4,5,6,7,8,9,10,11,12,13,14,15);
}

// B fragment: logical 32x16 (KxN) where the source is stored row-major as
// [N][K] (i.e. we compute X * W^T).  lane holds col n0+(lane&15),
// 16 contiguous K values at k0 + (lane>>4)*16.
__device__ __forceinline__ v16h load_BT(const _Float16* __restrict__ p, int ld,
                                        int n0, int k0, int lane) {
  const int n = n0 + (lane & 15);
  const int h = lane >> 4;
  return *(const v16h*)(p + (size_t)n * ld + k0 + h * 16);
}

// Async global -> LDS copy, 16 bytes per lane (tracked by ASYNCcnt).
__device__ __forceinline__ void async_b128(unsigned lds_byte_off,
                                           unsigned long long gaddr) {
  asm volatile("global_load_async_to_lds_b128 %0, %1, off"
               :: "v"(lds_byte_off), "v"(gaddr) : "memory");
}

// ---------------------------------------------------------------------------
// f32 -> f16 cast
// ---------------------------------------------------------------------------
__global__ void __launch_bounds__(256)
cast_f32_f16(const float* __restrict__ src, _Float16* __restrict__ dst, int n) {
  int i = blockIdx.x * blockDim.x + threadIdx.x;
  if (i < n) dst[i] = (_Float16)src[i];
}

// ---------------------------------------------------------------------------
// GEMM: out[M,N] = A[M,K] * W[N,K]^T   (f16 in, f16 or f32 out)
// Block = 4 waves = 64 rows x 128 cols.  The 128-col x 32-k W slice is
// staged into LDS with double-buffered async global->LDS loads and shared
// by all 4 waves (each wave owns a 16-row m-tile, 8 WMMA accumulators).
// ---------------------------------------------------------------------------
template <bool F32OUT>
__global__ void __launch_bounds__(128, 1)
gemm_xwT(const _Float16* __restrict__ A,
         const _Float16* __restrict__ W,
         _Float16* __restrict__ outH,
         float* __restrict__ outF,
         int M, int N, int K) {
  constexpr int ROWB  = 80;           // staged row stride in bytes (64 + pad)
  constexpr int TILEB = 128 * ROWB;   // one staged B tile: 10 KB
  __shared__ char smem[2 * TILEB];

  const int tilesN = N >> 7;
  const int bm  = blockIdx.x / tilesN;       // 64-row block index
  const int tn  = blockIdx.x % tilesN;
  const int wid  = threadIdx.x >> 5;
  const int lane = threadIdx.x & 31;
  const int half = lane >> 4;
  const int lcol = lane & 15;
  const int tid  = threadIdx.x;              // 0..127: stages W row n0+tid
  const int m0 = bm * 64 + wid * 16;
  const int n0 = tn << 7;

  const unsigned ldsBase = (unsigned)(unsigned long long)(&smem[0]);
  const _Float16* wrow = W + (size_t)(n0 + tid) * K;

  // prologue: stage k=0 slice into buffer 0
  {
    unsigned long long g = (unsigned long long)(wrow);
    unsigned dst = ldsBase + tid * ROWB;
#pragma unroll
    for (int c = 0; c < 4; ++c) async_b128(dst + c * 16, g + c * 16);
  }

  v8f acc[8] = {};
  const int nk = K >> 5;
  for (int jb = 0; jb < nk; ++jb) {
    const int k0  = jb << 5;
    const int cur = jb & 1;
    // issue next slice into the other buffer (last iter: harmless reload,
    // keeps the wait immediate constant; s_endpgm wait-idles the leftovers)
    const int kn = (jb + 1 < nk) ? (k0 + 32) : k0;
    {
      unsigned long long g = (unsigned long long)(wrow + kn);
      unsigned dst = ldsBase + (cur ^ 1) * TILEB + tid * ROWB;
#pragma unroll
      for (int c = 0; c < 4; ++c) async_b128(dst + c * 16, g + c * 16);
    }
    // in-order ASYNCcnt: <=4 outstanding  =>  current buffer's 4 landed
    asm volatile("s_wait_asynccnt 0x4" ::: "memory");
    __syncthreads();

    v16h a = load_A(A, K, m0, k0, lane);
    const char* bt = smem + cur * TILEB;
#pragma unroll
    for (int t = 0; t < 8; ++t) {
      const char* pp = bt + (16 * t + lcol) * ROWB + half * 32;
      v8h lo = *(const v8h*)(pp);
      v8h hi = *(const v8h*)(pp + 16);
      v16h b = __builtin_shufflevector(lo, hi,
                 0,1,2,3,4,5,6,7,8,9,10,11,12,13,14,15);
      acc[t] = wmma_f16(a, b, acc[t]);
    }
    __syncthreads();   // everyone done reading before next overwrite
  }

#pragma unroll
  for (int t = 0; t < 8; ++t) {
#pragma unroll
    for (int r = 0; r < 8; ++r) {
      const int row = m0 + r + 8 * half;
      const int col = n0 + 16 * t + lcol;
      if (F32OUT) outF[(size_t)row * N + col] = acc[t][r];
      else        outH[(size_t)row * N + col] = (_Float16)acc[t][r];
    }
  }
}

// ---------------------------------------------------------------------------
// RoPE on Q and K in place.  One thread per (b,l,h,j) pair, j in [0,64).
// ---------------------------------------------------------------------------
__global__ void __launch_bounds__(256)
rope_qk(_Float16* __restrict__ Qh, _Float16* __restrict__ Kh) {
  const int idx = blockIdx.x * blockDim.x + threadIdx.x;
  if (idx >= NBATCH * LSEQ * NHEAD * 64) return;
  const int j = idx & 63;
  const int h = (idx >> 6) & 15;
  const int l = (idx >> 10) & (LSEQ - 1);
  const int b = idx >> 21;
  const size_t base = ((size_t)(b * LSEQ + l)) * DHID + h * DHEAD;
  // inv_freq = theta^(-j/64) = exp(-j * ln(10000)/64)
  const float ang = (float)l * expf(-0.14391156831212787f * (float)j);
  const float c = cosf(ang), s = sinf(ang);
  float q1 = (float)Qh[base + j], q2 = (float)Qh[base + j + 64];
  Qh[base + j]      = (_Float16)(q1 * c - q2 * s);
  Qh[base + j + 64] = (_Float16)(q2 * c + q1 * s);
  float k1 = (float)Kh[base + j], k2 = (float)Kh[base + j + 64];
  Kh[base + j]      = (_Float16)(k1 * c - k2 * s);
  Kh[base + j + 64] = (_Float16)(k2 * c + k1 * s);
}

// ---------------------------------------------------------------------------
// V [b,l,h,d] -> VT [b,h,d,l]
// ---------------------------------------------------------------------------
__global__ void __launch_bounds__(256)
transpose_v(const _Float16* __restrict__ V, _Float16* __restrict__ VT) {
  const int idx = blockIdx.x * blockDim.x + threadIdx.x;
  if (idx >= NBATCH * LSEQ * DHID) return;
  const int d = idx & (DHEAD - 1);
  const int h = (idx >> 7) & (NHEAD - 1);
  const int l = (idx >> 11) & (LSEQ - 1);
  const int b = idx >> 22;
  VT[((size_t)((b * NHEAD + h) * DHEAD + d)) * LSEQ + l] = V[idx];
}

// ---------------------------------------------------------------------------
// Flash attention (causal).  One wave (block of 32) = 16 query rows of one
// (b,h).  Online softmax in f32; P routed through LDS for C->A relayout.
// ---------------------------------------------------------------------------
__device__ __forceinline__ float rmax16(float v) {
  v = fmaxf(v, __shfl_xor(v, 1, 32));
  v = fmaxf(v, __shfl_xor(v, 2, 32));
  v = fmaxf(v, __shfl_xor(v, 4, 32));
  v = fmaxf(v, __shfl_xor(v, 8, 32));
  return v;
}
__device__ __forceinline__ float rsum16(float v) {
  v += __shfl_xor(v, 1, 32);
  v += __shfl_xor(v, 2, 32);
  v += __shfl_xor(v, 4, 32);
  v += __shfl_xor(v, 8, 32);
  return v;
}

__global__ void __launch_bounds__(32, 1)
flash_attn(const _Float16* __restrict__ Q,
           const _Float16* __restrict__ Kc,
           const _Float16* __restrict__ VT,
           _Float16* __restrict__ O) {
  const int mtile = blockIdx.x & (LSEQ / 16 - 1);
  const int bh    = blockIdx.x >> 7;
  const int b = bh >> 4, h = bh & 15;
  const int lane = threadIdx.x & 31;
  const int half = lane >> 4;
  const int lcol = lane & 15;
  const int m0 = mtile << 4;

  const _Float16* Qp = Q  + (size_t)b * LSEQ * DHID + h * DHEAD;
  const _Float16* Kp = Kc + (size_t)b * LSEQ * DHID + h * DHEAD;
  const _Float16* Vp = VT + ((size_t)(b * NHEAD + h) * DHEAD) * LSEQ;

  v16h qa[4];
#pragma unroll
  for (int t = 0; t < 4; ++t) qa[t] = load_A(Qp, DHID, m0, 32 * t, lane);

  v8f acc[8] = {};
  float mrow[8], lrow[8];
#pragma unroll
  for (int r = 0; r < 8; ++r) { mrow[r] = -__builtin_inff(); lrow[r] = 0.f; }

  __shared__ _Float16 pS[16 * 32];
  const float scale = 0.08838834764831845f;  // 1/sqrt(128)

  const int nj = (m0 + 47) >> 5;             // 32-key blocks (causal bound)
  for (int jb = 0; jb < nj; ++jb) {
    const int j0 = jb << 5;
    v8f s0 = {}, s1 = {};
#pragma unroll
    for (int t = 0; t < 4; ++t) {
      v16h b0 = load_BT(Kp, DHID, j0,      32 * t, lane);
      v16h b1 = load_BT(Kp, DHID, j0 + 16, 32 * t, lane);
      s0 = wmma_f16(qa[t], b0, s0);
      s1 = wmma_f16(qa[t], b1, s1);
    }
    const int key0 = j0 + lcol, key1 = j0 + 16 + lcol;

    __syncthreads();   // single-wave WG: compiler fence before rewriting pS
#pragma unroll
    for (int r = 0; r < 8; ++r) {
      const int row = m0 + r + 8 * half;
      const float a0 = (key0 <= row) ? s0[r] * scale : -__builtin_inff();
      const float a1 = (key1 <= row) ? s1[r] * scale : -__builtin_inff();
      const float mn = fmaxf(mrow[r], rmax16(fmaxf(a0, a1)));
      const float e0 = __expf(a0 - mn);
      const float e1 = __expf(a1 - mn);
      const float alpha = __expf(mrow[r] - mn);
      lrow[r] = lrow[r] * alpha + rsum16(e0 + e1);
      mrow[r] = mn;
      pS[(r + 8 * half) * 32 + lcol]      = (_Float16)e0;
      pS[(r + 8 * half) * 32 + 16 + lcol] = (_Float16)e1;
#pragma unroll
      for (int db = 0; db < 8; ++db) acc[db][r] *= alpha;
    }
    __syncthreads();

    v16h pa;
#pragma unroll
    for (int i = 0; i < 8; ++i) {
      pa[i]     = pS[(lane & 15) * 32 + half * 16 + i];
      pa[8 + i] = pS[(lane & 15) * 32 + half * 16 + 8 + i];
    }
#pragma unroll
    for (int db = 0; db < 8; ++db) {
      v16h vb = load_BT(Vp, LSEQ, db * 16, j0, lane);
      acc[db] = wmma_f16(pa, vb, acc[db]);
    }
  }

#pragma unroll
  for (int r = 0; r < 8; ++r) {
    const float inv = 1.0f / lrow[r];
    const int row = m0 + r + 8 * half;
    _Float16* orow = O + ((size_t)(b * LSEQ + row)) * DHID + h * DHEAD;
#pragma unroll
    for (int db = 0; db < 8; ++db)
      orow[db * 16 + lcol] = (_Float16)(acc[db][r] * inv);
  }
}

// ---------------------------------------------------------------------------
// Host-side launch.  Workspace layout (f16 elements), total ~96 MB:
//   [0)        x16 (8M)   -> reused as VT after QKV GEMMs
//   [8388608)  wq16 (4M)  -> reused (with wk16) as O after flash
//   [12582912) wk16 (4M)
//   [16777216) wv16 (4M)
//   [20971520) wo16 (4M)
//   [25165824) q16 (8M)
//   [33554432) k16 (8M)
//   [41943040) v16 (8M)
// ---------------------------------------------------------------------------
extern "C" void kernel_launch(void* const* d_in, const int* in_sizes, int n_in,
                              void* d_out, int out_size, void* d_ws, size_t ws_size,
                              hipStream_t stream) {
  const float* x  = (const float*)d_in[0];
  const float* wq = (const float*)d_in[1];
  const float* wk = (const float*)d_in[2];
  const float* wv = (const float*)d_in[3];
  const float* wo = (const float*)d_in[4];
  float* out = (float*)d_out;

  _Float16* ws  = (_Float16*)d_ws;
  _Float16* X16  = ws;
  _Float16* WQ16 = ws + 8388608;
  _Float16* WK16 = ws + 12582912;
  _Float16* WV16 = ws + 16777216;
  _Float16* WO16 = ws + 20971520;
  _Float16* Q16  = ws + 25165824;
  _Float16* K16  = ws + 33554432;
  _Float16* V16  = ws + 41943040;
  _Float16* VT16 = X16;    // reuse (x dead after QKV GEMMs)
  _Float16* O16  = WQ16;   // reuse (wq/wk dead after their GEMMs)

  const int nX = MTOT * DHID;      // 8,388,608
  const int nW = DHID * DHID;      // 4,194,304

  cast_f32_f16<<<(nX + 255) / 256, 256, 0, stream>>>(x,  X16,  nX);
  cast_f32_f16<<<(nW + 255) / 256, 256, 0, stream>>>(wq, WQ16, nW);
  cast_f32_f16<<<(nW + 255) / 256, 256, 0, stream>>>(wk, WK16, nW);
  cast_f32_f16<<<(nW + 255) / 256, 256, 0, stream>>>(wv, WV16, nW);
  cast_f32_f16<<<(nW + 255) / 256, 256, 0, stream>>>(wo, WO16, nW);

  const int blocks = (MTOT / 64) * (DHID / 128);   // 1024 blocks of 128
  gemm_xwT<false><<<blocks, 128, 0, stream>>>(X16, WQ16, Q16, nullptr, MTOT, DHID, DHID);
  gemm_xwT<false><<<blocks, 128, 0, stream>>>(X16, WK16, K16, nullptr, MTOT, DHID, DHID);
  gemm_xwT<false><<<blocks, 128, 0, stream>>>(X16, WV16, V16, nullptr, MTOT, DHID, DHID);

  const int nR = NBATCH * LSEQ * NHEAD * 64;       // 4,194,304 rope pairs
  rope_qk<<<(nR + 255) / 256, 256, 0, stream>>>(Q16, K16);

  transpose_v<<<(nX + 255) / 256, 256, 0, stream>>>(V16, VT16);

  flash_attn<<<NBATCH * NHEAD * (LSEQ / 16), 32, 0, stream>>>(Q16, K16, VT16, O16);

  gemm_xwT<true><<<blocks, 128, 0, stream>>>(O16, WO16, nullptr, out, MTOT, DHID, DHID);
}
